// Encoder_WordLstm_12987981103308
// MI455X (gfx1250) — compile-verified, matrix operand order checked by reading, not run on hardware
//
#include <hip/hip_runtime.h>
#include <stdint.h>

// ---------------- problem dims ----------------
#define B_        64
#define S_        512
#define DC_       200       // embed dim (char == bichar)
#define IDIM      800       // (DC+DB)*2
#define H_        300       // HID == rnn hidden
#define KPAD      320       // padded K for HID/H dims (10 k-tiles of 32)
#define JPAD      304       // per-gate padded width (19 n-tiles of 16)
#define NG        1216      // 4*JPAD (76 n-tiles)
#define LDS_LD    328       // LDS row stride in halves (656B = 41*16B, bank-spread)

typedef __attribute__((ext_vector_type(16))) _Float16 v16h;
typedef __attribute__((ext_vector_type(8)))  float    v8f;

union FragU { v16h v; uint4 q[2]; };

// Load one 16x32 f16 WMMA operand fragment (A-style row layout; B uses the
// same per-lane pattern on W[N,K] rows since B is consumed column-major).
//   VGPR0..3 = K[k0 + hl*8 .. +7], VGPR4..7 = K[k0+16+hl*8 .. +7]
__device__ __forceinline__ v16h ldfrag(const _Float16* rowbase, int k0, int hl) {
  FragU f;
  f.q[0] = *reinterpret_cast<const uint4*>(rowbase + k0 + hl * 8);
  f.q[1] = *reinterpret_cast<const uint4*>(rowbase + k0 + 16 + hl * 8);
  return f.v;
}

__device__ __forceinline__ v8f wmma16(v16h a, v16h b, v8f c) {
  return __builtin_amdgcn_wmma_f32_16x16x32_f16(false, a, false, b, (short)0, c,
                                                false, false);
}

__device__ __forceinline__ float sigmoidf_(float x) {
  return 1.0f / (1.0f + __expf(-x));
}

// ---------------- weight staging ----------------
// W_lin [300,800] f32 -> [320,800] f16 (zero pad rows); bias -> [320] f32
__global__ void k_stage_wlin(const float* __restrict__ W, const float* __restrict__ b,
                             _Float16* __restrict__ Wst, float* __restrict__ bst) {
  int idx = blockIdx.x * blockDim.x + threadIdx.x;
  if (idx < KPAD * IDIM) {
    int n = idx / IDIM, k = idx - n * IDIM;
    float v = (n < H_) ? W[n * IDIM + k] : 0.0f;
    Wst[idx] = (_Float16)v;
  }
  if (idx < KPAD) bst[idx] = (idx < H_) ? b[idx] : 0.0f;
}

// Gate weights [1200,300] f32 -> [1216,320] f16 with per-gate 304 padding;
// optional bias [1200] -> [1216].
__global__ void k_stage_gate(const float* __restrict__ W, const float* __restrict__ b,
                             _Float16* __restrict__ Wst, float* __restrict__ bst) {
  int idx = blockIdx.x * blockDim.x + threadIdx.x;
  if (idx < NG * KPAD) {
    int n = idx / KPAD, k = idx - n * KPAD;
    int q = n / JPAD, j = n - q * JPAD;
    float v = 0.0f;
    if (j < H_ && k < H_) v = W[(q * H_ + j) * H_ + k];
    Wst[idx] = (_Float16)v;
  }
  if (bst != nullptr && idx < NG) {
    int q = idx / JPAD, j = idx - q * JPAD;
    bst[idx] = (j < H_) ? b[q * H_ + j] : 0.0f;
  }
}

// ---------------- embedding gather + concat -> f16 staging ----------------
__global__ void k_gather(const int* __restrict__ cf, const int* __restrict__ blf,
                         const int* __restrict__ brf, const int* __restrict__ scf,
                         const int* __restrict__ sblf, const int* __restrict__ sbrf,
                         const float* __restrict__ ce, const float* __restrict__ be,
                         const float* __restrict__ sce, const float* __restrict__ sbe,
                         _Float16* __restrict__ L, _Float16* __restrict__ R) {
  long idx = (long)blockIdx.x * blockDim.x + threadIdx.x;
  const long total = (long)B_ * S_ * IDIM;
  if (idx >= total) return;
  int  d = (int)(idx % IDIM);
  long m = idx / IDIM;                 // m = b*512 + s
  int seg = d / DC_, off = d - seg * DC_;
  float lv, rv;
  if (seg == 0)      { float v = ce [(long)cf [m] * DC_ + off]; lv = v; rv = v; }
  else if (seg == 1) { float v = sce[(long)scf[m] * DC_ + off]; lv = v; rv = v; }
  else if (seg == 2) { lv = be [(long)blf [m] * DC_ + off];
                       rv = be [(long)brf [m] * DC_ + off]; }
  else               { lv = sbe[(long)sblf[m] * DC_ + off];
                       rv = sbe[(long)sbrf[m] * DC_ + off]; }
  L[idx] = (_Float16)lv;
  R[idx] = (_Float16)rv;
}

// ---------------- GEMM1: act = tanh(A[32768x800] x W[320x800]^T + b) -> f16 [32768x320]
// 32x64 block per wave: 2 A-frags + 4 B-frags feed 8 WMMAs per k-step
// (64 acc VGPRs -- fits the register budget without spilling; launch_bounds
//  min-waves=1 lifts the compiler's 128-VGPR occupancy cap).
__global__ void __launch_bounds__(256, 1)
k_gemm_tanh(const _Float16* __restrict__ A, const _Float16* __restrict__ W,
            const float* __restrict__ bias, _Float16* __restrict__ out) {
  const int MTG = (B_ * S_) / 32;      // 1024 groups of 2 m-tiles
  const int NTG = KPAD / 64;           // 5 groups of 4 n-tiles
  int gw = (int)((blockIdx.x * blockDim.x + threadIdx.x) >> 5);
  if (gw >= MTG * NTG) return;
  int lane = threadIdx.x & 31, r = lane & 15, hl = lane >> 4;
  int mg = gw / NTG, ng = gw - mg * NTG;
  int m0 = mg * 32, n0 = ng * 64;
  v8f zero = {0.f,0.f,0.f,0.f,0.f,0.f,0.f,0.f};
  v8f acc[2][4];
#pragma unroll
  for (int mi = 0; mi < 2; ++mi)
#pragma unroll
    for (int t = 0; t < 4; ++t) acc[mi][t] = zero;
  for (int kt = 0; kt < IDIM / 32; ++kt) {
    int k0 = kt * 32;
    v16h a0 = ldfrag(A + (long)(m0 + r) * IDIM, k0, hl);
    v16h a1 = ldfrag(A + (long)(m0 + 16 + r) * IDIM, k0, hl);
#pragma unroll
    for (int t = 0; t < 4; ++t) {
      v16h bf = ldfrag(W + (long)(n0 + t * 16 + r) * IDIM, k0, hl);
      acc[0][t] = wmma16(a0, bf, acc[0][t]);
      acc[1][t] = wmma16(a1, bf, acc[1][t]);
    }
  }
#pragma unroll
  for (int t = 0; t < 4; ++t) {
    int n = n0 + t * 16 + r;           // D: col = lane&15
    float bn = bias[n];
#pragma unroll
    for (int mi = 0; mi < 2; ++mi)
#pragma unroll
      for (int v = 0; v < 8; ++v) {
        int m = m0 + mi * 16 + v + 8 * hl;   // D: row = VGPR + 8*(lane>>4)
        out[(long)m * KPAD + n] = (_Float16)tanhf(acc[mi][t][v] + bn);
      }
  }
}

// ---------------- GEMM2: X[s][b][n] = act[32768x320] x Wg[1216x320]^T + bg (f32) ----
__global__ void __launch_bounds__(256, 1)
k_gemm_xg(const _Float16* __restrict__ A, const _Float16* __restrict__ W,
          const float* __restrict__ bias, float* __restrict__ X) {
  const int MTG = (B_ * S_) / 32;      // 1024
  const int NTG = NG / 64;             // 19
  int gw = (int)((blockIdx.x * blockDim.x + threadIdx.x) >> 5);
  if (gw >= MTG * NTG) return;
  int lane = threadIdx.x & 31, r = lane & 15, hl = lane >> 4;
  int mg = gw / NTG, ng = gw - mg * NTG;
  int m0 = mg * 32, n0 = ng * 64;
  v8f zero = {0.f,0.f,0.f,0.f,0.f,0.f,0.f,0.f};
  v8f acc[2][4];
#pragma unroll
  for (int mi = 0; mi < 2; ++mi)
#pragma unroll
    for (int t = 0; t < 4; ++t) acc[mi][t] = zero;
  for (int kt = 0; kt < KPAD / 32; ++kt) {
    int k0 = kt * 32;
    v16h a0 = ldfrag(A + (long)(m0 + r) * KPAD, k0, hl);
    v16h a1 = ldfrag(A + (long)(m0 + 16 + r) * KPAD, k0, hl);
#pragma unroll
    for (int t = 0; t < 4; ++t) {
      v16h bf = ldfrag(W + (long)(n0 + t * 16 + r) * KPAD, k0, hl);
      acc[0][t] = wmma16(a0, bf, acc[0][t]);
      acc[1][t] = wmma16(a1, bf, acc[1][t]);
    }
  }
#pragma unroll
  for (int t = 0; t < 4; ++t) {
    int n = n0 + t * 16 + r;
    float bn = bias[n];
#pragma unroll
    for (int mi = 0; mi < 2; ++mi)
#pragma unroll
      for (int v = 0; v < 8; ++v) {
        int m = m0 + mi * 16 + v + 8 * hl;
        int s = m & (S_ - 1), b = m >> 9;            // m = b*512 + s
        X[((long)s * B_ + b) * NG + n] = acc[mi][t][v] + bn;
      }
  }
}

// ---------------- sequential LSTM scan, one block per direction ----------------
// 19 waves; wave w owns hidden slice j in [w*16, w*16+16). For each batch-tile it
// computes all 4 gate tiles so the cell update is register-local; c persists in
// registers across all 512 steps; h cycles through a 42KB LDS f16 buffer.
// Next step's 311KB X slice is prefetched (global_prefetch_b8) to hide HBM
// latency behind the current step's 3040 WMMAs.
__global__ void __launch_bounds__(608, 1)
k_lstm(const _Float16* __restrict__ WhhLs, const _Float16* __restrict__ WhhRs,
       const float* __restrict__ XL, const float* __restrict__ XR,
       float* __restrict__ out) {
  __shared__ _Float16 hbuf[B_ * LDS_LD];
  const int dir = blockIdx.x;
  const _Float16* W = dir ? WhhRs : WhhLs;
  const float*    X = dir ? XR : XL;
  const int tid = threadIdx.x;
  for (int i = tid; i < B_ * LDS_LD; i += blockDim.x) hbuf[i] = (_Float16)0.0f;
  __syncthreads();

  const int wave = tid >> 5, lane = tid & 31;
  const int r = lane & 15, hl = lane >> 4;
  const int j0 = wave * 16;
  const int j  = j0 + r;               // hidden index this lane's column maps to

  float c[4][8];
#pragma unroll
  for (int mt = 0; mt < 4; ++mt)
#pragma unroll
    for (int v = 0; v < 8; ++v) c[mt][v] = 0.0f;
  float hout[4][8];
  const v8f zero = {0.f,0.f,0.f,0.f,0.f,0.f,0.f,0.f};

  for (int t = 0; t < S_; ++t) {
    const int s = dir ? (S_ - 1 - t) : t;
    if (t + 1 < S_) {                   // prefetch next step's X slice (311,296 B)
      const int sn = dir ? (S_ - 2 - t) : (t + 1);
      const char* nx = (const char*)(X + (long)sn * B_ * NG) + (long)tid * 512;
      __builtin_prefetch(nx, 0, 1);
      __builtin_prefetch(nx + 256, 0, 1);
    }
#pragma unroll
    for (int mt = 0; mt < 4; ++mt) {
      const int m0 = mt * 16;
      v8f acc[4] = {zero, zero, zero, zero};
      for (int kt = 0; kt < KPAD / 32; ++kt) {
        const int k0 = kt * 32;
        FragU fa;                       // A fragment from LDS (ds_load_b128 x2)
        fa.q[0] = *reinterpret_cast<const uint4*>(&hbuf[(m0 + r) * LDS_LD + k0 + hl * 8]);
        fa.q[1] = *reinterpret_cast<const uint4*>(&hbuf[(m0 + r) * LDS_LD + k0 + 16 + hl * 8]);
#pragma unroll
        for (int q = 0; q < 4; ++q) {
          v16h bf = ldfrag(W + (long)(q * JPAD + j0 + r) * KPAD, k0, hl);
          acc[q] = wmma16(fa.v, bf, acc[q]);
        }
      }
#pragma unroll
      for (int v = 0; v < 8; ++v) {
        const int bb = m0 + v + 8 * hl;                  // batch index
        const long xb = ((long)s * B_ + bb) * NG + j0 + r;
        float pi = acc[0][v] + X[xb];
        float pf = acc[1][v] + X[xb + JPAD];
        float pg = acc[2][v] + X[xb + 2 * JPAD];
        float po = acc[3][v] + X[xb + 3 * JPAD];
        float ig = sigmoidf_(pi), fg = sigmoidf_(pf);
        float gg = tanhf(pg),     og = sigmoidf_(po);
        float cn = fg * c[mt][v] + ig * gg;
        c[mt][v] = cn;
        float h = og * tanhf(cn);
        hout[mt][v] = h;
        if (j < H_) out[((long)bb * S_ + s) * (2 * H_) + dir * H_ + j] = h;
      }
    }
    __syncthreads();                    // all reads of previous h done
#pragma unroll
    for (int mt = 0; mt < 4; ++mt)
#pragma unroll
      for (int v = 0; v < 8; ++v)
        hbuf[(mt * 16 + v + 8 * hl) * LDS_LD + j] = (_Float16)hout[mt][v];
    __syncthreads();                    // new h visible
  }
}

// ---------------- host side ----------------
extern "C" void kernel_launch(void* const* d_in, const int* in_sizes, int n_in,
                              void* d_out, int out_size, void* d_ws, size_t ws_size,
                              hipStream_t stream) {
  (void)in_sizes; (void)n_in; (void)out_size; (void)ws_size;
  const int*   cf   = (const int*)d_in[0];
  const int*   blf  = (const int*)d_in[1];
  const int*   brf  = (const int*)d_in[2];
  const int*   scf  = (const int*)d_in[3];
  const int*   sblf = (const int*)d_in[4];
  const int*   sbrf = (const int*)d_in[5];
  const float* ce   = (const float*)d_in[6];
  const float* be   = (const float*)d_in[7];
  const float* sce  = (const float*)d_in[8];
  const float* sbe  = (const float*)d_in[9];
  const float* Wlin = (const float*)d_in[10];
  const float* blin = (const float*)d_in[11];
  const float* WihL = (const float*)d_in[12];
  const float* WhhL = (const float*)d_in[13];
  const float* bLin_l = (const float*)d_in[14];
  const float* WihR = (const float*)d_in[15];
  const float* WhhR = (const float*)d_in[16];
  const float* bLin_r = (const float*)d_in[17];
  float* out = (float*)d_out;

  // workspace carve-up (peak ~364 MB; f16 concat staging aliases the later xl/xr)
  char* ws = (char*)d_ws;
  size_t off = 0;
  auto carve = [&](size_t bytes) -> char* {
    char* p = ws + off;
    off = (off + bytes + 255) & ~(size_t)255;
    return p;
  };
  const size_t M = (size_t)B_ * S_;
  float*    xl    = (float*)carve((size_t)S_ * B_ * NG * sizeof(float));     // 159.4 MB
  float*    xr    = (float*)carve((size_t)S_ * B_ * NG * sizeof(float));     // 159.4 MB
  _Float16* actL  = (_Float16*)carve(M * KPAD * sizeof(_Float16));           // 21 MB
  _Float16* actR  = (_Float16*)carve(M * KPAD * sizeof(_Float16));           // 21 MB
  _Float16* WlinS = (_Float16*)carve((size_t)KPAD * IDIM * sizeof(_Float16));
  float*    blinS = (float*)carve(KPAD * sizeof(float));
  _Float16* WihLS = (_Float16*)carve((size_t)NG * KPAD * sizeof(_Float16));
  _Float16* WihRS = (_Float16*)carve((size_t)NG * KPAD * sizeof(_Float16));
  _Float16* WhhLS = (_Float16*)carve((size_t)NG * KPAD * sizeof(_Float16));
  _Float16* WhhRS = (_Float16*)carve((size_t)NG * KPAD * sizeof(_Float16));
  float*    bLS   = (float*)carve(NG * sizeof(float));
  float*    bRS   = (float*)carve(NG * sizeof(float));
  // alias: f16 concat staging (52.4 MB each) lives inside xl/xr, which are
  // written only after the staging has been fully consumed by GEMM1.
  _Float16* Lbuf = (_Float16*)xl;
  _Float16* Rbuf = (_Float16*)xr;

  // 1) stage weights to padded f16
  {
    int n = KPAD * IDIM;
    k_stage_wlin<<<(n + 255) / 256, 256, 0, stream>>>(Wlin, blin, WlinS, blinS);
    n = NG * KPAD;
    k_stage_gate<<<(n + 255) / 256, 256, 0, stream>>>(WihL, bLin_l, WihLS, bLS);
    k_stage_gate<<<(n + 255) / 256, 256, 0, stream>>>(WihR, bLin_r, WihRS, bRS);
    k_stage_gate<<<(n + 255) / 256, 256, 0, stream>>>(WhhL, nullptr, WhhLS, nullptr);
    k_stage_gate<<<(n + 255) / 256, 256, 0, stream>>>(WhhR, nullptr, WhhRS, nullptr);
  }
  // 2) embedding gather + concat
  {
    long total = (long)M * IDIM;
    k_gather<<<(unsigned)((total + 255) / 256), 256, 0, stream>>>(
        cf, blf, brf, scf, sblf, sbrf, ce, be, sce, sbe, Lbuf, Rbuf);
  }
  // 3) input linear + tanh (WMMA, 32x64 per wave)
  {
    int waves = (int)(M / 32) * (KPAD / 64);          // 5120
    int blocks = (waves * 32 + 255) / 256;
    k_gemm_tanh<<<blocks, 256, 0, stream>>>(Lbuf, WlinS, blinS, actL);
    k_gemm_tanh<<<blocks, 256, 0, stream>>>(Rbuf, WlinS, blinS, actR);
  }
  // 4) gate precompute (WMMA), scattered to [s][b][gate] — overwrites Lbuf/Rbuf aliases
  {
    int waves = (int)(M / 32) * (NG / 64);            // 19456
    int blocks = (waves * 32 + 255) / 256;
    k_gemm_xg<<<blocks, 256, 0, stream>>>(actL, WihLS, bLS, xl);
    k_gemm_xg<<<blocks, 256, 0, stream>>>(actR, WihRS, bRS, xr);
  }
  // 5) bidirectional sequential scan: block 0 = L->R, block 1 = R->L
  k_lstm<<<2, 608, 0, stream>>>(WhhLS, WhhRS, xl, xr, out);
}
